// AttentionBlock_47691316855203
// MI455X (gfx1250) — matrix operand order repaired and checked by reading, not compiled, not run on hardware
//
#include <hip/hip_runtime.h>
#include <math.h>

#define S_HW   4096
#define C_CH   512
#define NBATCH 4
#define NGROUP 4
#define CPG    128          // channels per group
#define EPSV   1e-6f

typedef __attribute__((ext_vector_type(16))) __bf16 v16bf;
typedef __attribute__((ext_vector_type(8)))  float  v8f;

union FragAB { v16bf v; uint4 q[2]; };
union FragC  { v8f   v; float f[8]; };

// hardware f32 -> bf16 convert (RNE); avoids 3-4 integer VALU ops per element
__device__ __forceinline__ unsigned short f2bf(float f) {
  return __builtin_bit_cast(unsigned short, (__bf16)f);
}
__device__ __forceinline__ float bf2f(unsigned short h) {
  unsigned int u = ((unsigned int)h) << 16;
  return __builtin_bit_cast(float, u);
}
__device__ __forceinline__ unsigned int pk2(float a, float b) {
  return (unsigned int)f2bf(a) | ((unsigned int)f2bf(b) << 16);
}
__device__ __forceinline__ v8f zero8() {
  v8f z = {0.f,0.f,0.f,0.f,0.f,0.f,0.f,0.f}; return z;
}
__device__ __forceinline__ float ldf(const float* p)          { return *p; }
__device__ __forceinline__ float ldf(const unsigned short* p) { return bf2f(*p); }

__device__ __forceinline__ v8f wmma_bf16(const FragAB& a, const FragAB& b, v8f c) {
  return __builtin_amdgcn_wmma_f32_16x16x32_bf16(false, a.v, false, b.v,
                                                 (short)0, c, false, false);
}

// CDNA5 async copy: global -> LDS without touching VGPR data, ASYNCcnt-tracked.
// Generic pointers to __shared__ carry the LDS byte offset in their low 32 bits.
__device__ __forceinline__ void async_b128(void* lds_ptr, const void* gptr) {
  asm volatile("global_load_async_to_lds_b128 %0, %1, off"
               :: "v"((unsigned)(uintptr_t)lds_ptr),
                  "v"((unsigned long long)(uintptr_t)gptr)
               : "memory");
}
__device__ __forceinline__ void wait_async0() {
  asm volatile("s_wait_asynccnt 0x0" ::: "memory");
}

// scatter 8 bf16 (one contiguous global run along N) into LDS column kk
__device__ __forceinline__ void scatter8(unsigned short (*Bsb)[40], int n, int kk, uint4 d) {
  Bsb[n + 0][kk] = (unsigned short)(d.x);
  Bsb[n + 1][kk] = (unsigned short)(d.x >> 16);
  Bsb[n + 2][kk] = (unsigned short)(d.y);
  Bsb[n + 3][kk] = (unsigned short)(d.y >> 16);
  Bsb[n + 4][kk] = (unsigned short)(d.z);
  Bsb[n + 5][kk] = (unsigned short)(d.z >> 16);
  Bsb[n + 6][kk] = (unsigned short)(d.w);
  Bsb[n + 7][kk] = (unsigned short)(d.w >> 16);
}

// ---------------------------------------------------------------------------
// GroupNorm: stats (one block per (b,g)) then elementwise apply -> bf16
// ---------------------------------------------------------------------------
template<typename T>
__global__ __launch_bounds__(256) void gn_stats_kernel(const T* __restrict__ x,
                                                       float* __restrict__ stats) {
  const int bg = blockIdx.x;                           // b*NGROUP + g
  const size_t n = (size_t)CPG * S_HW;                 // 524288
  const T* p = x + (size_t)bg * n;
  float s = 0.f, ss = 0.f;
  for (size_t i = threadIdx.x; i < n; i += 256) {
    float v = ldf(p + i);
    s += v; ss += v * v;
  }
  __shared__ float r1[256], r2[256];
  r1[threadIdx.x] = s; r2[threadIdx.x] = ss;
  __syncthreads();
  for (int off = 128; off > 0; off >>= 1) {
    if ((int)threadIdx.x < off) {
      r1[threadIdx.x] += r1[threadIdx.x + off];
      r2[threadIdx.x] += r2[threadIdx.x + off];
    }
    __syncthreads();
  }
  if (threadIdx.x == 0) {
    float mean = r1[0] / (float)n;
    float var  = r2[0] / (float)n - mean * mean;
    stats[bg * 2 + 0] = mean;
    stats[bg * 2 + 1] = rsqrtf(var + EPSV);
  }
}

template<typename T>
__global__ __launch_bounds__(256) void gn_apply_kernel(const T* __restrict__ x,
    const float* __restrict__ stats, const float* __restrict__ gamma,
    const float* __restrict__ beta, unsigned short* __restrict__ y, int total) {
  int idx = blockIdx.x * 256 + threadIdx.x;
  if (idx >= total) return;
  int c  = (idx / S_HW) % C_CH;
  int b  =  idx / (S_HW * C_CH);
  int bg = b * NGROUP + c / CPG;
  float mean = stats[bg * 2 + 0];
  float rstd = stats[bg * 2 + 1];
  float v = ldf(x + idx);
  y[idx] = f2bf((v - mean) * rstd * gamma[c] + beta[c]);
}

// ---------------------------------------------------------------------------
// GEMM: A = fp32 weights (Mtot x K, converted to bf16 in staging),
//       B = bf16 activations ([b][K][Ntot]), 128x128 tile, kstep 32,
//       double-buffered LDS so global loads overlap WMMA.
// EPI: 0 = bf16 out [b][m][n], 1 = bf16 out transposed [b][n][m],
//      2 = fp32 out [b][m][n] + residual (final projection).
// ---------------------------------------------------------------------------
template<int EPI>
__global__ __launch_bounds__(256) void gemm_wf32_kernel(
    const float* __restrict__ A, const unsigned short* __restrict__ Bg,
    const float* __restrict__ bias, void* __restrict__ outp,
    const float* __restrict__ resid, int Mtot, int Ntot, int K) {
  __shared__ unsigned short As[2][128][40];   // [buf][m][k]
  __shared__ unsigned short Bs[2][128][40];   // [buf][n][k] (transposed)
  const int b    = blockIdx.z;
  const unsigned short* B = Bg + (size_t)b * K * Ntot;
  const int m0   = blockIdx.y * 128;
  const int n0   = blockIdx.x * 128;
  const int tid  = threadIdx.x;
  const int lane = tid & 31;
  const int wid  = tid >> 5;
  const int wm   = (wid & 3) * 32;
  const int wn   = (wid >> 2) * 64;

  v8f acc[2][4];
  #pragma unroll
  for (int i = 0; i < 2; ++i)
    #pragma unroll
    for (int j = 0; j < 4; ++j) acc[i][j] = zero8();

  const int arow = tid >> 1;               // 0..127
  const int acol = (tid & 1) * 16;         // 0 or 16
  const int brow = tid >> 4;               // 0..15
  const int bcol = (tid & 15) * 8;         // 0..120
  const float* aSrc = A + (size_t)(m0 + arow) * K + acol;
  const unsigned short* bSrc = B + n0 + bcol;

  // prologue: stage tile 0 into buffer 0
  {
    float4 f0 = ((const float4*)aSrc)[0];
    float4 f1 = ((const float4*)aSrc)[1];
    float4 f2 = ((const float4*)aSrc)[2];
    float4 f3 = ((const float4*)aSrc)[3];
    unsigned int* d32 = (unsigned int*)&As[0][arow][acol];
    d32[0] = pk2(f0.x, f0.y); d32[1] = pk2(f0.z, f0.w);
    d32[2] = pk2(f1.x, f1.y); d32[3] = pk2(f1.z, f1.w);
    d32[4] = pk2(f2.x, f2.y); d32[5] = pk2(f2.z, f2.w);
    d32[6] = pk2(f3.x, f3.y); d32[7] = pk2(f3.z, f3.w);
    scatter8(Bs[0], bcol, brow,      *(const uint4*)(bSrc + (size_t)brow * Ntot));
    scatter8(Bs[0], bcol, brow + 16, *(const uint4*)(bSrc + (size_t)(16 + brow) * Ntot));
  }
  __syncthreads();

  const int nk = K >> 5;
  for (int t = 0; t < nk; ++t) {
    const int cur = t & 1, nxt = cur ^ 1;
    const bool has = (t + 1) < nk;
    const int kn = (t + 1) * 32;
    float4 f0, f1, f2, f3; uint4 d0, d1;
    if (has) {          // issue next-tile global loads before compute
      f0 = ((const float4*)(aSrc + kn))[0];
      f1 = ((const float4*)(aSrc + kn))[1];
      f2 = ((const float4*)(aSrc + kn))[2];
      f3 = ((const float4*)(aSrc + kn))[3];
      d0 = *(const uint4*)(bSrc + (size_t)(kn + brow) * Ntot);
      d1 = *(const uint4*)(bSrc + (size_t)(kn + 16 + brow) * Ntot);
    }

    FragAB afr[2], bfr[4];
    const int mrow  = lane & 15;
    const int koffA = (lane >> 4) * 8;
    #pragma unroll
    for (int i = 0; i < 2; ++i) {
      const unsigned short* pa = &As[cur][wm + i * 16 + mrow][0];
      afr[i].q[0] = *(const uint4*)(pa + koffA);
      afr[i].q[1] = *(const uint4*)(pa + 16 + koffA);
    }
    const int koffB = (lane >> 4) * 8;
    #pragma unroll
    for (int j = 0; j < 4; ++j) {
      const unsigned short* pb = &Bs[cur][wn + j * 16 + (lane & 15)][0];
      bfr[j].q[0] = *(const uint4*)(pb + koffB);
      bfr[j].q[1] = *(const uint4*)(pb + 16 + koffB);
    }
    #pragma unroll
    for (int i = 0; i < 2; ++i)
      #pragma unroll
      for (int j = 0; j < 4; ++j)
        acc[i][j] = wmma_bf16(afr[i], bfr[j], acc[i][j]);

    if (has) {          // stage next tile into the other buffer
      unsigned int* d32 = (unsigned int*)&As[nxt][arow][acol];
      d32[0] = pk2(f0.x, f0.y); d32[1] = pk2(f0.z, f0.w);
      d32[2] = pk2(f1.x, f1.y); d32[3] = pk2(f1.z, f1.w);
      d32[4] = pk2(f2.x, f2.y); d32[5] = pk2(f2.z, f2.w);
      d32[6] = pk2(f3.x, f3.y); d32[7] = pk2(f3.z, f3.w);
      scatter8(Bs[nxt], bcol, brow,      d0);
      scatter8(Bs[nxt], bcol, brow + 16, d1);
    }
    __syncthreads();
  }

  const size_t bOff = (size_t)b * Mtot * Ntot;
  const int crow = lane >> 4, ccol = lane & 15;
  #pragma unroll
  for (int i = 0; i < 2; ++i)
    #pragma unroll
    for (int j = 0; j < 4; ++j) {
      FragC c; c.v = acc[i][j];
      #pragma unroll
      for (int r = 0; r < 8; ++r) {
        int gm = m0 + wm + i * 16 + crow * 8 + r;
        int gn = n0 + wn + j * 16 + ccol;
        float val = c.f[r] + bias[gm];
        if (EPI == 0) {
          ((unsigned short*)outp)[bOff + (size_t)gm * Ntot + gn] = f2bf(val);
        } else if (EPI == 1) {
          ((unsigned short*)outp)[bOff + (size_t)gn * Mtot + gm] = f2bf(val);
        } else {
          size_t o = bOff + (size_t)gm * Ntot + gn;
          ((float*)outp)[o] = val + resid[o];
        }
      }
    }
}

// ---------------------------------------------------------------------------
// GEMM: A bf16 (M x K row major), B bf16 (BT=0: [K][N], BT=1: [N][K]).
// Double-buffered; untransformed operands are staged with CDNA5 async
// global->LDS copies (ASYNCcnt). OUTF32 selects fp32 vs bf16 output.
// ---------------------------------------------------------------------------
template<int BT, int OUTF32>
__global__ __launch_bounds__(256) void gemm_bf16_kernel(
    const unsigned short* __restrict__ A, const unsigned short* __restrict__ B,
    void* __restrict__ outp, float alpha, int Mtot, int Ntot, int K) {
  __shared__ unsigned short As[2][128][40];
  __shared__ unsigned short Bs[2][128][40];   // [n][k]
  const int m0   = blockIdx.y * 128;
  const int n0   = blockIdx.x * 128;
  const int tid  = threadIdx.x;
  const int lane = tid & 31;
  const int wid  = tid >> 5;
  const int wm   = (wid & 3) * 32;
  const int wn   = (wid >> 2) * 64;

  v8f acc[2][4];
  #pragma unroll
  for (int i = 0; i < 2; ++i)
    #pragma unroll
    for (int j = 0; j < 4; ++j) acc[i][j] = zero8();

  const int arow = tid >> 1;
  const int acol = (tid & 1) * 16;
  const int brow = tid >> 4;
  const int bcol = (tid & 15) * 8;
  const unsigned short* aSrc  = A + (size_t)(m0 + arow) * K + acol;
  const unsigned short* bSrcT = B + (size_t)(n0 + arow) * K + acol;  // BT path
  const unsigned short* bSrc  = B + n0 + bcol;                       // !BT path

  // prologue: tile 0 -> buffer 0 (async where data is copied unmodified)
  async_b128(&As[0][arow][acol],     aSrc);
  async_b128(&As[0][arow][acol + 8], aSrc + 8);
  if (BT) {
    async_b128(&Bs[0][arow][acol],     bSrcT);
    async_b128(&Bs[0][arow][acol + 8], bSrcT + 8);
  } else {
    scatter8(Bs[0], bcol, brow,      *(const uint4*)(bSrc + (size_t)brow * Ntot));
    scatter8(Bs[0], bcol, brow + 16, *(const uint4*)(bSrc + (size_t)(16 + brow) * Ntot));
  }
  wait_async0();
  __syncthreads();

  const int nk = K >> 5;
  for (int t = 0; t < nk; ++t) {
    const int cur = t & 1, nxt = cur ^ 1;
    const bool has = (t + 1) < nk;
    const int kn = (t + 1) * 32;
    uint4 d0, d1;
    if (has) {
      async_b128(&As[nxt][arow][acol],     aSrc + kn);
      async_b128(&As[nxt][arow][acol + 8], aSrc + kn + 8);
      if (BT) {
        async_b128(&Bs[nxt][arow][acol],     bSrcT + kn);
        async_b128(&Bs[nxt][arow][acol + 8], bSrcT + kn + 8);
      } else {
        d0 = *(const uint4*)(bSrc + (size_t)(kn + brow) * Ntot);
        d1 = *(const uint4*)(bSrc + (size_t)(kn + 16 + brow) * Ntot);
      }
    }

    FragAB afr[2], bfr[4];
    const int mrow  = lane & 15;
    const int koffA = (lane >> 4) * 8;
    #pragma unroll
    for (int i = 0; i < 2; ++i) {
      const unsigned short* pa = &As[cur][wm + i * 16 + mrow][0];
      afr[i].q[0] = *(const uint4*)(pa + koffA);
      afr[i].q[1] = *(const uint4*)(pa + 16 + koffA);
    }
    const int koffB = (lane >> 4) * 8;
    #pragma unroll
    for (int j = 0; j < 4; ++j) {
      const unsigned short* pb = &Bs[cur][wn + j * 16 + (lane & 15)][0];
      bfr[j].q[0] = *(const uint4*)(pb + koffB);
      bfr[j].q[1] = *(const uint4*)(pb + 16 + koffB);
    }
    #pragma unroll
    for (int i = 0; i < 2; ++i)
      #pragma unroll
      for (int j = 0; j < 4; ++j)
        acc[i][j] = wmma_bf16(afr[i], bfr[j], acc[i][j]);

    if (has && !BT) {
      scatter8(Bs[nxt], bcol, brow,      d0);
      scatter8(Bs[nxt], bcol, brow + 16, d1);
    }
    wait_async0();
    __syncthreads();
  }

  const int crow = lane >> 4, ccol = lane & 15;
  #pragma unroll
  for (int i = 0; i < 2; ++i)
    #pragma unroll
    for (int j = 0; j < 4; ++j) {
      FragC c; c.v = acc[i][j];
      #pragma unroll
      for (int r = 0; r < 8; ++r) {
        int gm = m0 + wm + i * 16 + crow * 8 + r;
        int gn = n0 + wn + j * 16 + ccol;
        float val = alpha * c.f[r];
        if (OUTF32) ((float*)outp)[(size_t)gm * Ntot + gn] = val;
        else ((unsigned short*)outp)[(size_t)gm * Ntot + gn] = f2bf(val);
      }
    }
}

// ---------------------------------------------------------------------------
// Row softmax: fp32 scores (4096) -> bf16 probs. One block per row.
// ---------------------------------------------------------------------------
__global__ __launch_bounds__(256) void softmax_kernel(const float* __restrict__ Sm,
                                                      unsigned short* __restrict__ P) {
  const size_t row = blockIdx.x;
  const float* sr = Sm + row * S_HW;
  float loc[16];
  float mx = -1e30f;
  #pragma unroll
  for (int i = 0; i < 16; ++i) {
    loc[i] = sr[threadIdx.x + i * 256];
    mx = fmaxf(mx, loc[i]);
  }
  __shared__ float red[256];
  red[threadIdx.x] = mx; __syncthreads();
  for (int off = 128; off > 0; off >>= 1) {
    if ((int)threadIdx.x < off)
      red[threadIdx.x] = fmaxf(red[threadIdx.x], red[threadIdx.x + off]);
    __syncthreads();
  }
  mx = red[0]; __syncthreads();
  float s = 0.f;
  #pragma unroll
  for (int i = 0; i < 16; ++i) { loc[i] = __expf(loc[i] - mx); s += loc[i]; }
  red[threadIdx.x] = s; __syncthreads();
  for (int off = 128; off > 0; off >>= 1) {
    if ((int)threadIdx.x < off) red[threadIdx.x] += red[threadIdx.x + off];
    __syncthreads();
  }
  float inv = 1.0f / red[0];
  #pragma unroll
  for (int i = 0; i < 16; ++i)
    P[row * S_HW + threadIdx.x + i * 256] = f2bf(loc[i] * inv);
}

// ---------------------------------------------------------------------------
// Fused gated MLP GEMM: a = W1[0:C] @ m, g = W1[C:2C] @ m on the same tile,
// h = (a+b)*silu(g+b). W1 fp32 (2C x C).
// ---------------------------------------------------------------------------
__global__ __launch_bounds__(256) void mlp_gate_kernel(
    const float* __restrict__ W1, const float* __restrict__ b1,
    const unsigned short* __restrict__ Mg, unsigned short* __restrict__ H) {
  __shared__ unsigned short As[2][64][40];   // [set][m][k]
  __shared__ unsigned short Bs[128][40];     // [n][k]
  const int b  = blockIdx.z;
  const unsigned short* Bm = Mg + (size_t)b * C_CH * S_HW;
  const int m0 = blockIdx.y * 64;
  const int n0 = blockIdx.x * 128;
  const int tid  = threadIdx.x;
  const int lane = tid & 31;
  const int wid  = tid >> 5;
  const int wm   = (wid & 1) * 32;           // 2 waves along M(64)
  const int wn   = (wid >> 1) * 32;          // 4 waves along N(128)

  v8f acc[2][2][2];
  #pragma unroll
  for (int s = 0; s < 2; ++s)
    #pragma unroll
    for (int i = 0; i < 2; ++i)
      #pragma unroll
      for (int j = 0; j < 2; ++j) acc[s][i][j] = zero8();

  const int aset = tid >> 7;                 // 0: a-rows, 1: g-rows
  const int arow = (tid & 127) >> 1;         // 0..63
  const int acol = (tid & 1) * 16;
  const int brow = tid >> 4;
  const int bcol = (tid & 15) * 8;

  for (int k0 = 0; k0 < C_CH; k0 += 32) {
    {
      const float* src = W1 + (size_t)(aset * C_CH + m0 + arow) * C_CH + k0 + acol;
      float4 f0 = ((const float4*)src)[0];
      float4 f1 = ((const float4*)src)[1];
      float4 f2 = ((const float4*)src)[2];
      float4 f3 = ((const float4*)src)[3];
      unsigned int* d32 = (unsigned int*)&As[aset][arow][acol];
      d32[0] = pk2(f0.x, f0.y); d32[1] = pk2(f0.z, f0.w);
      d32[2] = pk2(f1.x, f1.y); d32[3] = pk2(f1.z, f1.w);
      d32[4] = pk2(f2.x, f2.y); d32[5] = pk2(f2.z, f2.w);
      d32[6] = pk2(f3.x, f3.y); d32[7] = pk2(f3.z, f3.w);
    }
    scatter8(Bs, bcol, brow,
             *(const uint4*)(Bm + (size_t)(k0 + brow) * S_HW + n0 + bcol));
    scatter8(Bs, bcol, brow + 16,
             *(const uint4*)(Bm + (size_t)(k0 + 16 + brow) * S_HW + n0 + bcol));
    __syncthreads();

    FragAB afr[2][2], bfr[2];
    const int mrow  = lane & 15;
    const int koffA = (lane >> 4) * 8;
    #pragma unroll
    for (int s = 0; s < 2; ++s)
      #pragma unroll
      for (int i = 0; i < 2; ++i) {
        const unsigned short* pa = &As[s][wm + i * 16 + mrow][0];
        afr[s][i].q[0] = *(const uint4*)(pa + koffA);
        afr[s][i].q[1] = *(const uint4*)(pa + 16 + koffA);
      }
    const int koffB = (lane >> 4) * 8;
    #pragma unroll
    for (int j = 0; j < 2; ++j) {
      const unsigned short* pb = &Bs[wn + j * 16 + (lane & 15)][0];
      bfr[j].q[0] = *(const uint4*)(pb + koffB);
      bfr[j].q[1] = *(const uint4*)(pb + 16 + koffB);
    }
    #pragma unroll
    for (int s = 0; s < 2; ++s)
      #pragma unroll
      for (int i = 0; i < 2; ++i)
        #pragma unroll
        for (int j = 0; j < 2; ++j)
          acc[s][i][j] = wmma_bf16(afr[s][i], bfr[j], acc[s][i][j]);
    __syncthreads();
  }

  const size_t bOff = (size_t)b * C_CH * S_HW;
  const int crow = lane >> 4, ccol = lane & 15;
  #pragma unroll
  for (int i = 0; i < 2; ++i)
    #pragma unroll
    for (int j = 0; j < 2; ++j) {
      FragC ca, cg;
      ca.v = acc[0][i][j]; cg.v = acc[1][i][j];
      #pragma unroll
      for (int r = 0; r < 8; ++r) {
        int gm = m0 + wm + i * 16 + crow * 8 + r;
        int gn = n0 + wn + j * 16 + ccol;
        float a = ca.f[r] + b1[gm];
        float g = cg.f[r] + b1[C_CH + gm];
        float h = a * (g / (1.0f + __expf(-g)));   // a * silu(g)
        H[bOff + (size_t)gm * S_HW + gn] = f2bf(h);
      }
    }
}

// ---------------------------------------------------------------------------
extern "C" void kernel_launch(void* const* d_in, const int* in_sizes, int n_in,
                              void* d_out, int out_size, void* d_ws, size_t ws_size,
                              hipStream_t stream) {
  (void)in_sizes; (void)n_in; (void)out_size; (void)ws_size;
  const float* x    = (const float*)d_in[0];
  const float* gns  = (const float*)d_in[1];
  const float* gnb  = (const float*)d_in[2];
  const float* qw   = (const float*)d_in[3];
  const float* qb   = (const float*)d_in[4];
  const float* kw   = (const float*)d_in[5];
  const float* kb   = (const float*)d_in[6];
  const float* vw   = (const float*)d_in[7];
  const float* vb   = (const float*)d_in[8];
  const float* gn2s = (const float*)d_in[9];
  const float* gn2b = (const float*)d_in[10];
  const float* w1   = (const float*)d_in[11];
  const float* b1   = (const float*)d_in[12];
  const float* w2   = (const float*)d_in[13];
  const float* b2   = (const float*)d_in[14];
  float* out = (float*)d_out;

  const size_t CS = (size_t)C_CH * S_HW;        // 2M elements
  unsigned short* xn   = (unsigned short*)d_ws;           // [b][c][s] bf16
  unsigned short* q    = xn   + (size_t)NBATCH * CS;      // [b][s][c] bf16
  unsigned short* kbf  = q    + (size_t)NBATCH * CS;      // [b][c][s] bf16
  unsigned short* v    = kbf  + (size_t)NBATCH * CS;      // [b][c][s] bf16
  unsigned short* ao   = v    + (size_t)NBATCH * CS;      // attn out [b][c][s]
  unsigned short* mn   = ao   + (size_t)NBATCH * CS;      // GN2 out
  unsigned short* hbuf = mn   + (size_t)NBATCH * CS;      // gated MLP hidden
  float* stats = (float*)(hbuf + (size_t)NBATCH * CS);    // 64 floats
  float* Sbuf  = stats + 64;                              // 4096x4096 fp32 (per-batch)
  unsigned short* P = (unsigned short*)(Sbuf + (size_t)S_HW * S_HW); // probs bf16

  const int total = NBATCH * C_CH * S_HW;
  const float scale = 0.044194173824159216f;    // 1/sqrt(512)

  // GroupNorm 1
  gn_stats_kernel<float><<<NBATCH * NGROUP, 256, 0, stream>>>(x, stats);
  gn_apply_kernel<float><<<total / 256, 256, 0, stream>>>(x, stats, gns, gnb, xn, total);

  // q/k/v projections (q stored transposed [s][c] for the QK^T A-operand)
  dim3 gp(S_HW / 128, C_CH / 128, NBATCH);
  gemm_wf32_kernel<1><<<gp, 256, 0, stream>>>(qw, xn, qb, q,   nullptr, C_CH, S_HW, C_CH);
  gemm_wf32_kernel<0><<<gp, 256, 0, stream>>>(kw, xn, kb, kbf, nullptr, C_CH, S_HW, C_CH);
  gemm_wf32_kernel<0><<<gp, 256, 0, stream>>>(vw, xn, vb, v,   nullptr, C_CH, S_HW, C_CH);

  // attention, batch at a time (score/prob buffers reused)
  for (int b = 0; b < NBATCH; ++b) {
    dim3 gs(S_HW / 128, S_HW / 128);
    gemm_bf16_kernel<0, 1><<<gs, 256, 0, stream>>>(q + (size_t)b * CS,
        kbf + (size_t)b * CS, Sbuf, scale, S_HW, S_HW, C_CH);
    softmax_kernel<<<S_HW, 256, 0, stream>>>(Sbuf, P);
    dim3 go(S_HW / 128, C_CH / 128);
    gemm_bf16_kernel<1, 0><<<go, 256, 0, stream>>>(v + (size_t)b * CS, P,
        ao + (size_t)b * CS, 1.0f, C_CH, S_HW, S_HW);
  }

  // GroupNorm 2 + gated MLP + residual
  gn_stats_kernel<unsigned short><<<NBATCH * NGROUP, 256, 0, stream>>>(ao, stats + 32);
  gn_apply_kernel<unsigned short><<<total / 256, 256, 0, stream>>>(ao, stats + 32,
      gn2s, gn2b, mn, total);
  dim3 g1(S_HW / 128, C_CH / 64, NBATCH);
  mlp_gate_kernel<<<g1, 256, 0, stream>>>(w1, b1, mn, hbuf);
  gemm_wf32_kernel<2><<<gp, 256, 0, stream>>>(w2, hbuf, b2, out, x, C_CH, S_HW, C_CH);
}
